// GraphAttentionLayer_2714419331845
// MI455X (gfx1250) — matrix operand order repaired and checked by reading, not compiled, not run on hardware
//
#include <hip/hip_runtime.h>
#include <hip/hip_bf16.h>

// ---------------------------------------------------------------------------
// GAT layer, fused flash-attention style, CDNA5 (gfx1250) wave32 WMMA bf16.
//   B=8, N=2048, F_IN=F_OUT=128
// Kernel 0: convert h and W to bf16 once (kills per-iteration v_cvt VALU).
// Kernel 1: Wh = h@W^T (WMMA bf16), e_i = Wh@a1, e_j = Wh@a2, store Wh^T bf16.
// Kernel 2: per 16-row i-tile, stream j in 32-wide tiles: adj tile staged to
//           LDS by the Tensor Data Mover (double buffered, s_wait_tensorcnt),
//           masked logits, online softmax, acc = acc*corr + alpha @ Wh (WMMA).
// ---------------------------------------------------------------------------

typedef __attribute__((ext_vector_type(16))) __bf16   v16bf;
typedef __attribute__((ext_vector_type(8)))  float    v8f;
typedef __attribute__((ext_vector_type(4)))  unsigned v4u;
typedef __attribute__((ext_vector_type(8)))  int      v8i;
typedef __attribute__((ext_vector_type(4)))  int      v4i;

static constexpr int Bb = 8;
static constexpr int Nn = 2048;
static constexpr int Ff = 128;

#if defined(__HIP_DEVICE_COMPILE__) && defined(__gfx1250__) && \
    __has_builtin(__builtin_amdgcn_tensor_load_to_lds) && \
    __has_builtin(__builtin_amdgcn_s_wait_tensorcnt)
#define USE_TDM 1
#else
#define USE_TDM 0
#endif

// K-offset pair base covered by A-fragment dword v (0..7) for half 'hi'
// (16-bit A-matrix 16x32 layout, CDNA5 ISA §7.12.2)
__device__ __forceinline__ int kslotA(int v, int hi) {
    return (v < 4) ? (hi * 8 + 2 * v) : (16 + hi * 8 + 2 * (v - 4));
}
// K-offset pair base for B-fragment dword v (B 32x16 bf16 layout)
__device__ __forceinline__ int kslotB(int v, int hi) {
    return (v < 4) ? (hi * 8 + 2 * v) : (16 + hi * 8 + 2 * (v - 4));
}

__device__ __forceinline__ unsigned pack_bf16(float x, float y) {
    const unsigned lo = (unsigned)__builtin_bit_cast(unsigned short, (__bf16)x);
    const unsigned hh = (unsigned)__builtin_bit_cast(unsigned short, (__bf16)y);
    return lo | (hh << 16);
}

#if USE_TDM
// Issue a TDM load of a 16-row x 32-col f32 tile (row stride Nn) into LDS.
// D# layout per CDNA5 ISA (2-D tensor, groups 2/3 zero).
// This toolchain's builtin takes 6 args: (g0 v4u, g1 v8i, g2 v4i, g3 v4i, v8i, cpol).
__device__ __forceinline__ void tdm_load_adj_tile(const float* gptr, unsigned lds_off) {
    const unsigned long long ga = (unsigned long long)(const void*)gptr;
    v4u g0;
    g0[0] = 1u;                                            // count=1, user mode
    g0[1] = lds_off;                                       // lds_addr (bytes)
    g0[2] = (unsigned)(ga & 0xFFFFFFFFu);                  // global_addr[31:0]
    g0[3] = (unsigned)((ga >> 32) & 0x01FFFFFFu) | (2u << 30); // addr[56:32], type=2
    v8i g1;
    g1[0] = (int)(2u << 16);                               // wg_mask=0, data_size=4B
    g1[1] = (int)((unsigned)Nn << 16);                     // tensor_dim0[15:0]
    g1[2] = (int)((unsigned)Nn << 16);                     // dim0 hi=0 | tensor_dim1[15:0]
    g1[3] = (int)(32u << 16);                              // dim1 hi=0 | tile_dim0=32
    g1[4] = 16;                                            // tile_dim1=16, tile_dim2=0
    g1[5] = Nn;                                            // tensor_dim0_stride[31:0]
    g1[6] = 0;                                             // stride0 hi | stride1 lo
    g1[7] = 0;
    v4i gz4 = {0, 0, 0, 0};
    v8i gz8 = {0, 0, 0, 0, 0, 0, 0, 0};
    __builtin_amdgcn_tensor_load_to_lds(g0, g1, gz4, gz4, gz8, 0);
}
#endif

// ---------------------------------------------------------------------------
// Kernel 0: one-time f32 -> bf16 conversion of h and W (pair-packed dwords).
// ---------------------------------------------------------------------------
__global__ __launch_bounds__(256)
void cvt_bf16_kernel(const float* __restrict__ h, const float* __restrict__ W,
                     unsigned* __restrict__ hbf, unsigned* __restrict__ wbf)
{
    const long NPH = (long)Bb * Nn * Ff / 2;   // h bf16-pair count
    const long NPW = (long)Ff * Ff / 2;        // W bf16-pair count
    const long idx = (long)blockIdx.x * blockDim.x + threadIdx.x;
    if (idx < NPH) {
        hbf[idx] = pack_bf16(h[2 * idx], h[2 * idx + 1]);
    } else if (idx < NPH + NPW) {
        const long k = idx - NPH;
        wbf[k] = pack_bf16(W[2 * k], W[2 * k + 1]);
    }
}

// ---------------------------------------------------------------------------
// Kernel 1: one wave per 16-row tile of flattened (B*N, F) h.
//   A fragment = 2x b128 from hbf, B fragment = 2x b128 from wbf (no cvt).
// ---------------------------------------------------------------------------
__global__ __launch_bounds__(32)
void wh_ei_ej_kernel(const unsigned* __restrict__ hbf, const unsigned* __restrict__ wbf,
                     const float* __restrict__ a, __bf16* __restrict__ wh_t,
                     float* __restrict__ ei, float* __restrict__ ej)
{
    const int t    = blockIdx.x;          // 16-row tile index in [0, B*N/16)
    const int lane = threadIdx.x;
    const int m    = lane & 15;
    const int hi   = lane >> 4;
    const long row = (long)16 * t + m;    // A-matrix row handled by this lane

    v8f zero = {};
    v8f acc[8];
#pragma unroll
    for (int ot = 0; ot < 8; ++ot) acc[ot] = zero;

#pragma unroll
    for (int k0 = 0; k0 < Ff; k0 += 32) {
        // A fragment: dwords 0..3 cover K = k0+hi*8..+7, dwords 4..7 cover +16
        union { v4u q[2]; v16bf w; } af;
        af.q[0] = *(const v4u*)&hbf[(row * Ff + k0 + hi * 8) >> 1];
        af.q[1] = *(const v4u*)&hbf[(row * Ff + k0 + 16 + hi * 8) >> 1];
#pragma unroll
        for (int ot = 0; ot < 8; ++ot) {
            const int n = 16 * ot + m;    // output feature column for this lane
            // B fragment: K = k0+hi*8..+7 and k0+16+hi*8..+7 at fixed n = W row n (row-major)
            union { v4u q[2]; v16bf w; } bfr;
            const int kb = k0 + hi * 8;
            bfr.q[0] = *(const v4u*)&wbf[(n * Ff + kb) >> 1];
            bfr.q[1] = *(const v4u*)&wbf[(n * Ff + kb + 16) >> 1];
            acc[ot] = __builtin_amdgcn_wmma_f32_16x16x32_bf16(
                false, af.w, false, bfr.w, (short)0, acc[ot], false, false);
        }
    }

    // C/D layout: lane holds rows (r + 8*hi), column (16*ot + m)
    const int b  = (16 * t) / Nn;
    const int jb = (16 * t) % Nn;

    float eip[8], ejp[8];
#pragma unroll
    for (int r = 0; r < 8; ++r) { eip[r] = 0.0f; ejp[r] = 0.0f; }

    unsigned* wt32 = (unsigned*)wh_t;
#pragma unroll
    for (int ot = 0; ot < 8; ++ot) {
        const int o = 16 * ot + m;
        const float a1 = a[o];
        const float a2 = a[Ff + o];
#pragma unroll
        for (int r = 0; r < 8; ++r) {
            const float v = acc[ot][r];
            eip[r] += v * a1;
            ejp[r] += v * a2;
        }
        // store Wh^T[b][o][j] as bf16, two consecutive rows packed per dword
#pragma unroll
        for (int r = 0; r < 8; r += 2) {
            const long idx = (long)(b * Ff + o) * Nn + jb + r + 8 * hi; // even
            wt32[idx >> 1] = pack_bf16(acc[ot][r], acc[ot][r + 1]);
        }
    }

    // reduce e_i/e_j over the 16 feature-column lanes of each half
#pragma unroll
    for (int r = 0; r < 8; ++r) {
#pragma unroll
        for (int mask = 1; mask < 16; mask <<= 1) {
            eip[r] += __shfl_xor(eip[r], mask, 32);
            ejp[r] += __shfl_xor(ejp[r], mask, 32);
        }
    }
    if (m == 0) {
#pragma unroll
        for (int r = 0; r < 8; ++r) {
            const int rr = 16 * t + r + 8 * hi;
            ei[rr] = eip[r];
            ej[rr] = ejp[r];
        }
    }
}

// ---------------------------------------------------------------------------
// Kernel 2: one wave per (batch, 16 i-rows); flash-style stream over j.
//   adj tiles staged to LDS via Tensor Data Mover (double buffered) when
//   available; otherwise direct b64 global loads.
// ---------------------------------------------------------------------------
__global__ __launch_bounds__(32)
void gat_softmax_av_kernel(const float* __restrict__ adj,
                           const __bf16* __restrict__ wh_t,
                           const float* __restrict__ ei,
                           const float* __restrict__ ej,
                           float* __restrict__ out)
{
    const int b    = blockIdx.y;
    const int i0   = blockIdx.x * 16;
    const int lane = threadIdx.x;
    const int m    = lane & 15;
    const int hi   = lane >> 4;

    const float ei_m = ei[b * Nn + i0 + m];                     // A row = m
    const float* ej_b    = ej + b * Nn;
    const unsigned* wt32 = (const unsigned*)wh_t;

    float mrun = -1.0e30f;
    float srun = 0.0f;
    v8f zero = {};
    v8f acc[8];
#pragma unroll
    for (int ot = 0; ot < 8; ++ot) acc[ot] = zero;

#if USE_TDM
    __shared__ float adj_tile[2][16 * 32];
    const float* adj_base = adj + (long)(b * Nn + i0) * Nn;     // wave-uniform
    const unsigned lds0 = (unsigned)(unsigned long long)(const void*)&adj_tile[0][0];
    const unsigned lds1 = (unsigned)(unsigned long long)(const void*)&adj_tile[1][0];
    tdm_load_adj_tile(adj_base + 0,  lds0);                     // tile 0
    tdm_load_adj_tile(adj_base + 32, lds1);                     // tile 1
#else
    const float* adj_row = adj + (long)(b * Nn + i0 + m) * Nn;
#endif

    for (int j0 = 0; j0 < Nn; j0 += 32) {
        // masked attention logits at exactly this lane's A-fragment slots
        float e[16];
#if USE_TDM
        __builtin_amdgcn_s_wait_tensorcnt(1);                   // tile (j0/32) done
        const float* tile = &adj_tile[(j0 >> 5) & 1][0];
#pragma unroll
        for (int vv = 0; vv < 8; ++vv) {
            const int kk  = kslotA(vv, hi);
            const float ad0 = tile[m * 32 + kk];
            const float ad1 = tile[m * 32 + kk + 1];
            const float ej0 = ej_b[j0 + kk];
            const float ej1 = ej_b[j0 + kk + 1];
            e[2 * vv]     = ei_m + ej0 + (1.0f - ad0) * -1.0e9f;
            e[2 * vv + 1] = ei_m + ej1 + (1.0f - ad1) * -1.0e9f;
        }
#else
#pragma unroll
        for (int vv = 0; vv < 8; ++vv) {
            const int kk = kslotA(vv, hi);
            const int j  = j0 + kk;
            const float ad0 = adj_row[j];
            const float ad1 = adj_row[j + 1];
            const float ej0 = ej_b[j];
            const float ej1 = ej_b[j + 1];
            e[2 * vv]     = ei_m + ej0 + (1.0f - ad0) * -1.0e9f;
            e[2 * vv + 1] = ei_m + ej1 + (1.0f - ad1) * -1.0e9f;
        }
        if (j0 + 32 < Nn) __builtin_prefetch(&adj_row[j0 + 32 + hi * 16], 0, 1);
#endif

        // online softmax: row m lives on lanes {m, m+16}
        float tmax = e[0];
#pragma unroll
        for (int s = 1; s < 16; ++s) tmax = fmaxf(tmax, e[s]);
        tmax = fmaxf(tmax, __shfl_xor(tmax, 16, 32));
        const float mnew  = fmaxf(mrun, tmax);
        const float scale = __expf(mrun - mnew);

        float alpha[16];
        float tsum = 0.0f;
#pragma unroll
        for (int s = 0; s < 16; ++s) {
            alpha[s] = __expf(e[s] - mnew);
            tsum += alpha[s];
        }
        tsum += __shfl_xor(tsum, 16, 32);
        srun = srun * scale + tsum;
        mrun = mnew;

        // rescale accumulators: acc row (r + 8*hi)'s scale is owned by that lane
        float sc[8];
#pragma unroll
        for (int r = 0; r < 8; ++r) sc[r] = __shfl(scale, r + 8 * hi, 32);
#pragma unroll
        for (int ot = 0; ot < 8; ++ot) {
#pragma unroll
            for (int r = 0; r < 8; ++r) acc[ot][r] *= sc[r];
        }

        // A fragment (alpha, bf16)
        v16bf af;
#pragma unroll
        for (int s = 0; s < 16; ++s) af[s] = (__bf16)alpha[s];

#if USE_TDM
        // refill the buffer we just consumed with tile (j0/32 + 2)
        if (j0 + 64 < Nn) {
            asm volatile("s_wait_dscnt 0x0" ::: "memory");      // LDS reads retired
            tdm_load_adj_tile(adj_base + j0 + 64,
                              ((j0 >> 5) & 1) ? lds1 : lds0);
        }
#endif

        // 8 output tiles: B fragment = Wh^T[b][16*ot+m][j0 + kslotB] bf16 pairs
#pragma unroll
        for (int ot = 0; ot < 8; ++ot) {
            const int o = 16 * ot + m;
            union { unsigned u[8]; v16bf w; } bfr;
#pragma unroll
            for (int vv = 0; vv < 8; ++vv) {
                const long idx = (long)(b * Ff + o) * Nn + j0 + kslotB(vv, hi);
                bfr.u[vv] = wt32[idx >> 1];
            }
            acc[ot] = __builtin_amdgcn_wmma_f32_16x16x32_bf16(
                false, af, false, bfr.w, (short)0, acc[ot], false, false);
        }
    }

    // epilogue: divide by running softmax sum, store coalesced f32
    const float inv = 1.0f / srun;                 // owned for row m
    float invr[8];
#pragma unroll
    for (int r = 0; r < 8; ++r) invr[r] = __shfl(inv, r + 8 * hi, 32);
#pragma unroll
    for (int ot = 0; ot < 8; ++ot) {
#pragma unroll
        for (int r = 0; r < 8; ++r) {
            const long oidx = (long)(b * Nn + i0 + r + 8 * hi) * Ff + 16 * ot + m;
            out[oidx] = acc[ot][r] * invr[r];
        }
    }
}

// ---------------------------------------------------------------------------
// Host launch
// ---------------------------------------------------------------------------
extern "C" void kernel_launch(void* const* d_in, const int* in_sizes, int n_in,
                              void* d_out, int out_size, void* d_ws, size_t ws_size,
                              hipStream_t stream) {
    const float* h   = (const float*)d_in[0];   // (8, 2048, 128)
    const float* adj = (const float*)d_in[1];   // (8, 2048, 2048)
    const float* W   = (const float*)d_in[2];   // (128, 128)
    const float* a   = (const float*)d_in[3];   // (1, 256)
    float* out = (float*)d_out;                 // (8, 2048, 128)

    char* ws = (char*)d_ws;
    size_t off = 0;
    __bf16* wh_t = (__bf16*)(ws + off);  off += (size_t)Bb * Ff * Nn * 2;  // 4 MB
    float*  ei   = (float*)(ws + off);   off += (size_t)Bb * Nn * 4;       // 64 KB
    float*  ej   = (float*)(ws + off);   off += (size_t)Bb * Nn * 4;       // 64 KB
    unsigned* hbf = (unsigned*)(ws + off); off += (size_t)Bb * Nn * Ff * 2; // 4 MB
    unsigned* wbf = (unsigned*)(ws + off); off += (size_t)Ff * Ff * 2;      // 32 KB

    const long npairs = (long)Bb * Nn * Ff / 2 + (long)Ff * Ff / 2;
    cvt_bf16_kernel<<<dim3((unsigned)((npairs + 255) / 256)), dim3(256), 0, stream>>>(
        h, W, hbf, wbf);
    wh_ei_ej_kernel<<<dim3(Bb * Nn / 16), dim3(32), 0, stream>>>(hbf, wbf, a, wh_t, ei, ej);
    gat_softmax_av_kernel<<<dim3(Nn / 16, Bb), dim3(32), 0, stream>>>(adj, wh_t, ei, ej, out);
}
